// encoder_26431228739593
// MI455X (gfx1250) — matrix operand order repaired and checked by reading, not compiled, not run on hardware
//
#include <hip/hip_runtime.h>
#include <hip/hip_bf16.h>
#include <math.h>

// ---------------------------------------------------------------------------
// Nystrom-attention encoder + sparse neighbor aggregation, CDNA5 (gfx1250).
// All GEMMs use v_wmma_f32_16x16x32_bf16 (fp32 global data, bf16 staged in
// LDS, fp32 accumulate), 128x128x32 tiles, double-buffered LDS pipeline.
// wave32 everywhere.
// ---------------------------------------------------------------------------

typedef __attribute__((ext_vector_type(16))) __bf16 bf16x16;
typedef __attribute__((ext_vector_type(8)))  float  f32x8;

#define HEADS   8
#define DHEAD   64
#define MLAND   256
#define DMODEL  1024
#define INNER   512
#define QKVN    1536
#define CONVK   33

// ------------------------------ GEMM ---------------------------------------
constexpr int BM = 128, BN = 128, BK = 32;
constexpr int AROWU32 = (BK + 2) / 2;  // 17 dwords / A row (bank-conflict pad)
constexpr int BROWU32 = BN + 2;        // 130 dwords / B k-pair row

__device__ __forceinline__ unsigned pkbf(float a, float b) {
  union { __bf16 h[2]; unsigned u; } x;
  x.h[0] = (__bf16)a; x.h[1] = (__bf16)b;
  return x.u;
}

// C[b] = alpha * A[b] @ op(B[b]) (+bias[col]) (+res[row,col])
// A rows < rowPad read as zero (front sequence padding). Requires K%32==0.
__launch_bounds__(256)
__global__ void gemm_bf16_kernel(const float* __restrict__ A, int lda, long long sA, int rowPad,
                                 const float* __restrict__ B, int ldb, long long sB, int transB,
                                 float* __restrict__ C, int ldc, long long sC,
                                 const float* __restrict__ bias,
                                 const float* __restrict__ res, int ldres,
                                 float alpha, int M, int N, int K) {
  __shared__ unsigned As[2][BM * AROWU32];
  __shared__ unsigned Bs[2][(BK / 2) * BROWU32];

  const int b = blockIdx.z;
  A += (long long)b * sA;  B += (long long)b * sB;  C += (long long)b * sC;

  const int bm = blockIdx.y * BM, bn = blockIdx.x * BN;
  const int tid = threadIdx.x, lane = tid & 31, wave = tid >> 5;
  const int wr = (wave & 3) * 32;        // 4 row groups of 32
  const int wc = (wave >> 2) * 64;       // 2 col groups of 64 (4 n-tiles each)
  const int lh = lane & 15, l16 = lane >> 4;
  const int Msrc = M - rowPad;

  f32x8 acc[2][4];
#pragma unroll
  for (int i = 0; i < 2; ++i)
#pragma unroll
    for (int j = 0; j < 4; ++j)
#pragma unroll
      for (int e = 0; e < 8; ++e) acc[i][j][e] = 0.f;

  float4 ar[4], br[4];  // in-flight staging registers (next tile)

  // ---- load tile (global -> VGPR) ----------------------------------------
  auto loadTiles = [&](int k0) {
#pragma unroll
    for (int i = 0; i < 4; ++i) {
      int idx = tid + i * 256;
      int r = idx >> 3, c4 = (idx & 7) * 4;  // 128 rows x 8 float4
      int gr = bm + r - rowPad;
      ar[i] = make_float4(0.f, 0.f, 0.f, 0.f);
      if ((unsigned)gr < (unsigned)Msrc)
        ar[i] = *(const float4*)(A + (long long)gr * lda + (k0 + c4));
    }
    if (!transB) {
#pragma unroll
      for (int i = 0; i < 4; ++i) {
        int idx = tid + i * 256;
        int kk = idx >> 5, n4 = (idx & 31) * 4;  // 32 k-rows x 32 float4
        br[i] = make_float4(0.f, 0.f, 0.f, 0.f);
        if (bn + n4 < N)
          br[i] = *(const float4*)(B + (long long)(k0 + kk) * ldb + (bn + n4));
      }
    } else {  // B is N x K row-major; op(B)=B^T
#pragma unroll
      for (int i = 0; i < 4; ++i) {
        int idx = tid + i * 256;
        int nn = idx >> 3, k4 = (idx & 7) * 4;   // 128 n-rows x 8 float4
        br[i] = make_float4(0.f, 0.f, 0.f, 0.f);
        if (bn + nn < N)
          br[i] = *(const float4*)(B + (long long)(bn + nn) * ldb + (k0 + k4));
      }
    }
  };

  // ---- store tile (VGPR -> LDS, fp32 -> bf16 pairs) ----------------------
  auto storeTiles = [&](int buf) {
    unsigned* as = As[buf];
    unsigned* bs = Bs[buf];
#pragma unroll
    for (int i = 0; i < 4; ++i) {
      int idx = tid + i * 256;
      int r = idx >> 3, c4 = (idx & 7) * 4;
      as[r * AROWU32 + (c4 >> 1) + 0] = pkbf(ar[i].x, ar[i].y);
      as[r * AROWU32 + (c4 >> 1) + 1] = pkbf(ar[i].z, ar[i].w);
    }
    if (!transB) {
      __bf16* bh = (__bf16*)bs;
#pragma unroll
      for (int i = 0; i < 4; ++i) {
        int idx = tid + i * 256;
        int kk = idx >> 5, n4 = (idx & 31) * 4;
        int prow = kk >> 1, hi = kk & 1;
        bh[2 * (prow * BROWU32 + n4 + 0) + hi] = (__bf16)br[i].x;
        bh[2 * (prow * BROWU32 + n4 + 1) + hi] = (__bf16)br[i].y;
        bh[2 * (prow * BROWU32 + n4 + 2) + hi] = (__bf16)br[i].z;
        bh[2 * (prow * BROWU32 + n4 + 3) + hi] = (__bf16)br[i].w;
      }
    } else {
#pragma unroll
      for (int i = 0; i < 4; ++i) {
        int idx = tid + i * 256;
        int nn = idx >> 3, k4 = (idx & 7) * 4;
        bs[((k4 >> 1) + 0) * BROWU32 + nn] = pkbf(br[i].x, br[i].y);
        bs[((k4 >> 1) + 1) * BROWU32 + nn] = pkbf(br[i].z, br[i].w);
      }
    }
  };

  // ---- fragments per documented CDNA5 layouts + 8 WMMAs ------------------
  auto compute = [&](int buf) {
    const unsigned* as = As[buf];
    const unsigned* bs = Bs[buf];
    union AF { bf16x16 v; unsigned u[8]; };
    AF af[2];
#pragma unroll
    for (int i = 0; i < 2; ++i) {
      int m = wr + i * 16 + lh;
      int kb2 = l16 * 4;  // lanes 0-15: K 0..7/16..23, lanes 16-31: K 8..15/24..31
#pragma unroll
      for (int j = 0; j < 4; ++j) {
        af[i].u[j]     = as[m * AROWU32 + kb2 + j];
        af[i].u[4 + j] = as[m * AROWU32 + 8 + kb2 + j];
      }
    }
#pragma unroll
    for (int j2 = 0; j2 < 4; ++j2) {
      AF bfr;
      int nn = wc + j2 * 16 + lh;
      int p0 = l16 * 4;  // lanes 0-15: K 0..7/16..23, lanes 16-31: K 8..15/24..31
#pragma unroll
      for (int j = 0; j < 8; ++j) bfr.u[j] = bs[(p0 + (j & 3) + ((j >> 2) << 3)) * BROWU32 + nn];
#pragma unroll
      for (int i = 0; i < 2; ++i)
        acc[i][j2] = __builtin_amdgcn_wmma_f32_16x16x32_bf16(
            false, af[i].v, false, bfr.v, (short)0, acc[i][j2], false, false);
    }
  };

  // ---- double-buffered main loop -----------------------------------------
  loadTiles(0);
  storeTiles(0);
  __syncthreads();
  int cur = 0;
  for (int k0 = 0; k0 < K; k0 += BK) {
    const bool hasNext = (k0 + BK) < K;
    if (hasNext) loadTiles(k0 + BK);   // global loads in flight during WMMAs
    compute(cur);
    if (hasNext) storeTiles(cur ^ 1);
    __syncthreads();
    cur ^= 1;
  }

  // ---- epilogue: scale, bias, residual ------------------------------------
#pragma unroll
  for (int i = 0; i < 2; ++i)
#pragma unroll
    for (int j2 = 0; j2 < 4; ++j2)
#pragma unroll
      for (int r = 0; r < 8; ++r) {
        int row = bm + wr + i * 16 + r + l16 * 8;
        int col = bn + wc + j2 * 16 + lh;
        if (row < M && col < N) {
          float v = alpha * acc[i][j2][r];
          if (bias) v += bias[col];
          if (res)  v += res[(long long)row * ldres + col];
          C[(long long)row * ldc + col] = v;
        }
      }
}

// --------------------------- small kernels ---------------------------------
__global__ void zero_f32(float* p, int cnt) {
  int i = blockIdx.x * 256 + threadIdx.x;
  if (i < cnt) p[i] = 0.f;
}

__global__ void copy_f32(const float* __restrict__ s, float* __restrict__ d, int cnt) {
  int i = blockIdx.x * 256 + threadIdx.x;
  if (i < cnt) d[i] = s[i];
}

// landmark means over l consecutive rows of q and k (columns of qkv buffer)
__global__ void landmarks_kernel(const float* __restrict__ qkv,
                                 float* __restrict__ ql, float* __restrict__ kl,
                                 int l) {
  int j = blockIdx.x, h = blockIdx.y, d = threadIdx.x;
  long long base = (long long)j * l * QKVN + h * DHEAD + d;
  float sq = 0.f, sk = 0.f;
  for (int r = 0; r < l; ++r) {
    sq += qkv[base + (long long)r * QKVN];
    sk += qkv[base + (long long)r * QKVN + INNER];
  }
  float inv = 1.f / (float)l;
  ql[(h * MLAND + j) * DHEAD + d] = sq * inv;
  kl[(h * MLAND + j) * DHEAD + d] = sk * inv;
}

__global__ void softmax_rows(float* __restrict__ d, int width) {
  __shared__ float red[256];
  const int tid = threadIdx.x;
  float* row = d + (long long)blockIdx.x * width;
  float m = -3.402823466e38f;
  for (int i = tid; i < width; i += 256) m = fmaxf(m, row[i]);
  red[tid] = m; __syncthreads();
  for (int s = 128; s > 0; s >>= 1) { if (tid < s) red[tid] = fmaxf(red[tid], red[tid + s]); __syncthreads(); }
  const float mx = red[0]; __syncthreads();
  float sum = 0.f;
  for (int i = tid; i < width; i += 256) { float e = __expf(row[i] - mx); row[i] = e; sum += e; }
  red[tid] = sum; __syncthreads();
  for (int s = 128; s > 0; s >>= 1) { if (tid < s) red[tid] += red[tid + s]; __syncthreads(); }
  const float inv = 1.f / red[0];
  for (int i = tid; i < width; i += 256) row[i] *= inv;
}

// denom = max(rowsum|x|) * max(colsum|x|) over ALL heads (global, as in ref)
__global__ void pinv_prep(const float* __restrict__ a, float* scal) {
  __shared__ float red[256];
  const int h = blockIdx.y, i = blockIdx.x, tid = threadIdx.x;
  const float* A = a + (size_t)h * MLAND * MLAND;
  float rv = fabsf(A[i * MLAND + tid]);
  float cv = fabsf(A[tid * MLAND + i]);
  red[tid] = rv; __syncthreads();
  for (int s = 128; s > 0; s >>= 1) { if (tid < s) red[tid] += red[tid + s]; __syncthreads(); }
  float rowsum = red[0]; __syncthreads();
  red[tid] = cv; __syncthreads();
  for (int s = 128; s > 0; s >>= 1) { if (tid < s) red[tid] += red[tid + s]; __syncthreads(); }
  float colsum = red[0];
  if (tid == 0) {  // sums are >=0 -> uint compare == float compare
    atomicMax((unsigned int*)&scal[0], __float_as_uint(rowsum));
    atomicMax((unsigned int*)&scal[1], __float_as_uint(colsum));
  }
}

__global__ void pinv_init(const float* __restrict__ a, const float* __restrict__ scal,
                          float* __restrict__ z) {
  const int h = blockIdx.y, i = blockIdx.x, j = threadIdx.x;
  const float denom = scal[0] * scal[1];
  z[(size_t)h * MLAND * MLAND + i * MLAND + j] =
      a[(size_t)h * MLAND * MLAND + j * MLAND + i] / denom;
}

__global__ void ai_minus(float aval, const float* __restrict__ x, float* __restrict__ y) {
  int idx = blockIdx.x * 256 + threadIdx.x;  // 8*256*256 elements
  int i = (idx >> 8) & 255, j = idx & 255;
  float v = -x[idx];
  if (i == j) v += aval;
  y[idx] = v;
}

// depthwise conv along sequence on v, added into per-head encoder output
__global__ void conv_res(const float* __restrict__ qkv, const float* __restrict__ w,
                         float* __restrict__ enc512, int npad) {
  int i = blockIdx.x, h = blockIdx.y, d = threadIdx.x;
  float s = 0.f;
  for (int t = 0; t < CONVK; ++t) {
    int r = i + t - CONVK / 2;
    if ((unsigned)r < (unsigned)npad)
      s += qkv[(long long)r * QKVN + 2 * INNER + h * DHEAD + d] * w[h * CONVK + t];
  }
  enc512[(long long)i * INNER + h * DHEAD + d] += s;
}

// one wave per edge: 256-dim dot, wave reduce, segment-sum via atomicAdd
__global__ void edge_scores(const float* __restrict__ qp, const float* __restrict__ kp,
                            const int* __restrict__ ei, const float* __restrict__ av,
                            float* __restrict__ Araw, int E) {
  int e = blockIdx.x * 8 + (threadIdx.x >> 5);
  if (e >= E) return;
  int lane = threadIdx.x & 31;
  int src = ei[e], dst = ei[E + e];
  const float4* q4 = (const float4*)(qp + (long long)src * 256);
  const float4* k4 = (const float4*)(kp + (long long)dst * 256);
  float acc = 0.f;
  float4 a = q4[lane * 2],     b = k4[lane * 2];
  acc += a.x * b.x + a.y * b.y + a.z * b.z + a.w * b.w;
  a = q4[lane * 2 + 1];        b = k4[lane * 2 + 1];
  acc += a.x * b.x + a.y * b.y + a.z * b.z + a.w * b.w;
  for (int off = 16; off > 0; off >>= 1) acc += __shfl_down(acc, off, 32);
  if (lane == 0) atomicAdd(&Araw[src], acc * (1.f / 16.f) * av[e]);
}

__global__ void softmax_global(const float* __restrict__ a, float* __restrict__ alpha, int n) {
  __shared__ float red[1024];
  int tid = threadIdx.x;
  float m = -3.402823466e38f;
  for (int i = tid; i < n; i += 1024) m = fmaxf(m, a[i]);
  red[tid] = m; __syncthreads();
  for (int s = 512; s > 0; s >>= 1) { if (tid < s) red[tid] = fmaxf(red[tid], red[tid + s]); __syncthreads(); }
  float mx = red[0]; __syncthreads();
  float sum = 0.f;
  for (int i = tid; i < n; i += 1024) sum += __expf(a[i] - mx);
  red[tid] = sum; __syncthreads();
  for (int s = 512; s > 0; s >>= 1) { if (tid < s) red[tid] += red[tid + s]; __syncthreads(); }
  float inv = 1.f / red[0];
  for (int i = tid; i < n; i += 1024) alpha[i] = __expf(a[i] - mx) * inv;
}

__global__ void finalize_kernel(const float* __restrict__ value, const float* __restrict__ xenc,
                                const float* __restrict__ alpha, float* __restrict__ out, int n) {
  int idx = blockIdx.x * 256 + threadIdx.x;
  if (idx >= n * DMODEL) return;
  int row = idx >> 10;
  float xl = alpha[row] * value[idx];
  float sg = 1.f / (1.f + __expf(xl));   // sigmoid(-xl)
  float sw = sg * sg;
  out[idx] = 2.f * xl * sw + 2.f * xenc[idx] * (1.f - sw);
}

// ------------------------------- host --------------------------------------
extern "C" void kernel_launch(void* const* d_in, const int* in_sizes, int n_in,
                              void* d_out, int out_size, void* d_ws, size_t ws_size,
                              hipStream_t stream) {
  const float* dense  = (const float*)d_in[0];
  const int*   eidx   = (const int*)  d_in[1];
  const float* adjv   = (const float*)d_in[2];
  const float* qkv_w  = (const float*)d_in[3];
  const float* out_w  = (const float*)d_in[4];
  const float* out_b  = (const float*)d_in[5];
  const float* conv_w = (const float*)d_in[6];
  const float* wq_w   = (const float*)d_in[7];
  const float* wq_b   = (const float*)d_in[8];
  const float* wk_w   = (const float*)d_in[9];
  const float* wk_b   = (const float*)d_in[10];
  const float* wv_w   = (const float*)d_in[11];
  const float* wv_b   = (const float*)d_in[12];

  const int n    = in_sizes[0] / DMODEL;              // 30000
  const int E    = in_sizes[2];                       // 480000
  const int pad  = (MLAND - n % MLAND) % MLAND;       // 208
  const int npad = n + pad;                           // 30208
  const int l    = npad / MLAND;                      // 118
  const long long mm = (long long)MLAND * MLAND;      // 65536

  float* ws = (float*)d_ws;
  size_t cur = 0;
  auto alloc = [&](size_t cnt) { float* p = ws + cur; cur += cnt; return p; };
  float* qkv    = alloc((size_t)npad * QKVN);
  float* ql     = alloc((size_t)HEADS * MLAND * DHEAD);
  float* kl     = alloc((size_t)HEADS * MLAND * DHEAD);
  float* attn2  = alloc((size_t)HEADS * mm);
  float* z0     = alloc((size_t)HEADS * mm);
  float* z1     = alloc((size_t)HEADS * mm);
  float* XZ     = alloc((size_t)HEADS * mm);
  float* T      = alloc((size_t)HEADS * mm);
  float* T2     = alloc((size_t)HEADS * mm);
  float* a3v    = alloc((size_t)HEADS * MLAND * DHEAD);
  float* attn1  = alloc((size_t)HEADS * npad * MLAND);
  float* attn3  = alloc((size_t)HEADS * MLAND * npad);  // later reused for attn1@pinv
  float* enc512 = alloc((size_t)npad * INNER);
  float* xenc   = alloc((size_t)n * DMODEL);
  float* qp     = alloc((size_t)n * MLAND);
  float* kp     = alloc((size_t)n * MLAND);
  float* value  = alloc((size_t)n * DMODEL);
  float* nalpha = alloc((size_t)n);
  float* Araw   = alloc((size_t)n + 2);
  float* scal   = Araw + n;
  (void)ws_size; (void)n_in; (void)out_size;

  auto gemm = [&](const float* A, int lda, long long sA, int rowPad,
                  const float* B, int ldb, long long sB, bool trB,
                  float* C, int ldc, long long sC,
                  const float* bias, const float* res, int ldres,
                  float alpha, int M, int N, int K, int batch) {
    dim3 g((N + BN - 1) / BN, (M + BM - 1) / BM, batch);
    gemm_bf16_kernel<<<g, 256, 0, stream>>>(A, lda, sA, rowPad, B, ldb, sB, trB ? 1 : 0,
                                            C, ldc, sC, bias, res, ldres, alpha, M, N, K);
  };

  // zero accumulators (segment-sum + pinv global maxes) every launch
  zero_f32<<<(n + 2 + 255) / 256, 256, 0, stream>>>(Araw, n + 2);

  // ---- Nystrom encoder ----------------------------------------------------
  // qkv = pad(x) @ qkv_w   (front rows zero via rowPad)
  gemm(dense, DMODEL, 0, pad, qkv_w, QKVN, 0, false,
       qkv, QKVN, 0, nullptr, nullptr, 0, 1.f, npad, QKVN, DMODEL, 1);
  landmarks_kernel<<<dim3(MLAND, HEADS), DHEAD, 0, stream>>>(qkv, ql, kl, l);

  const float qscale = 0.125f;  // DIM_HEAD^-0.5
  // attn1 = softmax(q @ kl^T)  : per-head batched via column-slice strides
  gemm(qkv, QKVN, 64, 0, kl, DHEAD, (long long)MLAND * DHEAD, true,
       attn1, MLAND, (long long)npad * MLAND, nullptr, nullptr, 0,
       qscale, npad, MLAND, DHEAD, HEADS);
  softmax_rows<<<HEADS * npad, 256, 0, stream>>>(attn1, MLAND);
  // attn2 = softmax(ql @ kl^T)
  gemm(ql, DHEAD, (long long)MLAND * DHEAD, 0, kl, DHEAD, (long long)MLAND * DHEAD, true,
       attn2, MLAND, mm, nullptr, nullptr, 0, qscale, MLAND, MLAND, DHEAD, HEADS);
  softmax_rows<<<HEADS * MLAND, 256, 0, stream>>>(attn2, MLAND);
  // attn3 = softmax(ql @ k^T)
  gemm(ql, DHEAD, (long long)MLAND * DHEAD, 0, qkv + INNER, QKVN, 64, true,
       attn3, npad, (long long)MLAND * npad, nullptr, nullptr, 0,
       qscale, MLAND, npad, DHEAD, HEADS);
  softmax_rows<<<HEADS * MLAND, 256, 0, stream>>>(attn3, npad);

  // ---- iterative Moore-Penrose pinv of attn2 ------------------------------
  pinv_prep<<<dim3(MLAND, HEADS), 256, 0, stream>>>(attn2, scal);
  pinv_init<<<dim3(MLAND, HEADS), 256, 0, stream>>>(attn2, scal, z0);
  float* zc = z0; float* zn = z1;
  const int aiBlocks = (HEADS * (int)mm) / 256;
  for (int it = 0; it < 6; ++it) {
    gemm(attn2, MLAND, mm, 0, zc, MLAND, mm, false, XZ, MLAND, mm,
         nullptr, nullptr, 0, 1.f, MLAND, MLAND, MLAND, HEADS);
    ai_minus<<<aiBlocks, 256, 0, stream>>>(7.f, XZ, T);
    gemm(XZ, MLAND, mm, 0, T, MLAND, mm, false, T2, MLAND, mm,
         nullptr, nullptr, 0, 1.f, MLAND, MLAND, MLAND, HEADS);
    ai_minus<<<aiBlocks, 256, 0, stream>>>(15.f, T2, T);
    gemm(XZ, MLAND, mm, 0, T, MLAND, mm, false, T2, MLAND, mm,
         nullptr, nullptr, 0, 1.f, MLAND, MLAND, MLAND, HEADS);
    ai_minus<<<aiBlocks, 256, 0, stream>>>(13.f, T2, T);
    gemm(zc, MLAND, mm, 0, T, MLAND, mm, false, zn, MLAND, mm,
         nullptr, nullptr, 0, 0.25f, MLAND, MLAND, MLAND, HEADS);
    float* tmp = zc; zc = zn; zn = tmp;
  }

  // a3v = attn3 @ v
  gemm(attn3, npad, (long long)MLAND * npad, 0, qkv + 2 * INNER, QKVN, 64, false,
       a3v, DHEAD, (long long)MLAND * DHEAD, nullptr, nullptr, 0,
       1.f, MLAND, DHEAD, npad, HEADS);
  // attn1p = attn1 @ pinv  (reuse attn3 buffer)
  float* attn1p = attn3;
  gemm(attn1, MLAND, (long long)npad * MLAND, 0, zc, MLAND, mm, false,
       attn1p, MLAND, (long long)npad * MLAND, nullptr, nullptr, 0,
       1.f, npad, MLAND, MLAND, HEADS);
  // enc512[:, h*64:(h+1)*64] = attn1p @ a3v
  gemm(attn1p, MLAND, (long long)npad * MLAND, 0, a3v, DHEAD, (long long)MLAND * DHEAD, false,
       enc512, INNER, 64, nullptr, nullptr, 0, 1.f, npad, DHEAD, MLAND, HEADS);
  conv_res<<<dim3(npad, HEADS), DHEAD, 0, stream>>>(qkv, conv_w, enc512, npad);
  // xenc = enc512[pad:] @ out_w + out_b + dense   (drop front padding)
  gemm(enc512 + (size_t)pad * INNER, INNER, 0, 0, out_w, DMODEL, 0, false,
       xenc, DMODEL, 0, out_b, dense, DMODEL, 1.f, n, DMODEL, INNER, 1);

  // ---- sparse neighbor aggregation ---------------------------------------
  gemm(xenc, DMODEL, 0, 0, wq_w, MLAND, 0, false, qp, MLAND, 0,
       wq_b, nullptr, 0, 1.f, n, MLAND, DMODEL, 1);
  gemm(xenc, DMODEL, 0, 0, wk_w, MLAND, 0, false, kp, MLAND, 0,
       wk_b, nullptr, 0, 1.f, n, MLAND, DMODEL, 1);
  gemm(dense, DMODEL, 0, 0, wv_w, DMODEL, 0, false, value, DMODEL, 0,
       wv_b, nullptr, 0, 1.f, n, DMODEL, DMODEL, 1);

  edge_scores<<<(E + 7) / 8, 256, 0, stream>>>(qp, kp, eidx, adjv, Araw, E);
  softmax_global<<<1, 1024, 0, stream>>>(Araw, nalpha, n);

  float* out = (float*)d_out;
  finalize_kernel<<<(n * DMODEL + 255) / 256, 256, 0, stream>>>(value, xenc, nalpha, out, n);
  copy_f32<<<(n + 255) / 256, 256, 0, stream>>>(Araw, out + (size_t)n * DMODEL, n);
}
